// LaneSetAttentionEmbedder_64166811402256
// MI455X (gfx1250) — matrix operand order repaired and checked by reading, not compiled, not run on hardware
//
#include <hip/hip_runtime.h>

typedef __attribute__((ext_vector_type(16))) _Float16 v16h;
typedef __attribute__((ext_vector_type(8)))  float    v8f;

#define LN_EPS 1e-5f

constexpr int L_   = 1000000;
constexpr int F_   = 32;
constexpr int E_   = 64;
constexpr int K_   = 4;
constexpr int HID_ = 64;
constexpr int OUT_ = 64;

constexpr int WAVES_PER_BLOCK = 8;
constexpr int ROWS_PER_WAVE   = 16;
constexpr int ROWS_PER_BLOCK  = WAVES_PER_BLOCK * ROWS_PER_WAVE;            // 128
constexpr int NTILES          = L_ / ROWS_PER_WAVE;                         // 62500
constexpr int NBLK            = (L_ + ROWS_PER_BLOCK - 1) / ROWS_PER_BLOCK; // 7813

// workspace layout (float offsets)
constexpr int O_UQ    = 0;                    // K*E = 256 floats
constexpr int O_DEN   = 256;                  // K floats
constexpr int O_Z     = 320;                  // K*E = 256 floats
constexpr int O_HALF  = 640;                  // f16 weight copies: 10240 halves = 5120 floats
constexpr int H_W1    = 0;                    // half offsets inside the f16 region
constexpr int H_W2    = E_ * F_;              // 2048
constexpr int H_WW    = H_W2 + E_ * E_;       // 6144
constexpr int O_BPART = O_HALF + 5120;        // K*NBLK floats
constexpr int O_ZPART = O_BPART + K_ * NBLK;  // K*E*NBLK floats (~8 MB total ws)

// ---------------------------------------------------------------- helpers ---

__device__ __forceinline__ float fast_tanh(float x) {
#if __has_builtin(__builtin_amdgcn_tanhf)
  return __builtin_amdgcn_tanhf(x);        // native V_TANH_F32 (TRANS op)
#elif __has_builtin(__builtin_amdgcn_tanh_f32)
  return __builtin_amdgcn_tanh_f32(x);
#else
  return tanhf(x);
#endif
}

__device__ __forceinline__ v8f wmma_f16(v16h a, v16h b, v8f c) {
  // D = A(16x32 f16) x B(32x16 f16) + C(16x16 f32)
  return __builtin_amdgcn_wmma_f32_16x16x32_f16(false, a, false, b, (short)0, c,
                                                false, false);
}

// 16 consecutive f16 -> v16h (B-operand chunk; 32-byte aligned)
__device__ __forceinline__ v16h loadB(const _Float16* __restrict__ p) {
  return *reinterpret_cast<const v16h*>(p);
}

// A-operand for X (16x32 f16 from f32 memory):
// halves 0..15 -> K = kb..kb+15, kb = hi?16:0.
__device__ __forceinline__ v16h loadA_X(const float* __restrict__ X, int row, int hi) {
  v16h a;
  const float4* q = reinterpret_cast<const float4*>(X + row * F_ + (hi ? 16 : 0));
  float4 f0 = q[0], f1 = q[1], f2 = q[2], f3 = q[3];
  a[0] = (_Float16)f0.x;  a[1] = (_Float16)f0.y;  a[2] = (_Float16)f0.z;  a[3] = (_Float16)f0.w;
  a[4] = (_Float16)f1.x;  a[5] = (_Float16)f1.y;  a[6] = (_Float16)f1.z;  a[7] = (_Float16)f1.w;
  a[8] = (_Float16)f2.x;  a[9] = (_Float16)f2.y;  a[10] = (_Float16)f2.z; a[11] = (_Float16)f2.w;
  a[12] = (_Float16)f3.x; a[13] = (_Float16)f3.y; a[14] = (_Float16)f3.z; a[15] = (_Float16)f3.w;
  return a;
}

// A-operand reload from an LDS f16 tile [16 rows x 64 K], K-block kk (0/1)
__device__ __forceinline__ v16h loadA_lds(const _Float16* __restrict__ t, int m,
                                          int kk, int hi) {
  v16h a;
  const _Float16* p = t + m * E_ + kk * 32 + (hi ? 16 : 0);
#pragma unroll
  for (int h = 0; h < 8; ++h) {
    a[h]     = p[h];        // K = base + 0..7
    a[8 + h] = p[8 + h];    // K = base + 8..15
  }
  return a;
}

__device__ __forceinline__ float half_reduce(float v) {
  // sum over the 16 lanes sharing a row-group (masks stay within each half)
  v += __shfl_xor(v, 1, 32);
  v += __shfl_xor(v, 2, 32);
  v += __shfl_xor(v, 4, 32);
  v += __shfl_xor(v, 8, 32);
  return v;
}

// ---------------------------------------------------------------- kernels ---

// One-time prep: Uq = seeds @ Uw^T, plus f16 copies of w1/w2/Ww for direct
// v16h B-operand loads (kills the per-tile v_cvt_pk_f16_f32 storm).
__global__ __launch_bounds__(256) void prep_kernel(
    const float* __restrict__ seeds, const float* __restrict__ Uw,
    const float* __restrict__ w1, const float* __restrict__ w2,
    const float* __restrict__ Ww, float* __restrict__ uq,
    _Float16* __restrict__ hbase) {
  int t = threadIdx.x;
  if (t < E_) {
#pragma unroll
    for (int k = 0; k < K_; ++k) {
      float s = 0.f;
      for (int j = 0; j < E_; ++j) s += seeds[k * E_ + j] * Uw[t * E_ + j];
      uq[k * E_ + t] = s;
    }
  }
  for (int i = t; i < E_ * F_; i += 256) hbase[H_W1 + i] = (_Float16)w1[i];
  for (int i = t; i < E_ * E_; i += 256) hbase[H_W2 + i] = (_Float16)w2[i];
  for (int i = t; i < E_ * E_; i += 256) hbase[H_WW + i] = (_Float16)Ww[i];
}

// Pass 1: per-tile phi -> LN -> Wh -> scores -> per-block sums of exp(score)
__global__ __launch_bounds__(256) void pass1_kernel(
    const float* __restrict__ X, const _Float16* __restrict__ hbase,
    const float* __restrict__ b1, const float* __restrict__ b2,
    const float* __restrict__ gam, const float* __restrict__ bet,
    const float* __restrict__ Wb, const float* __restrict__ vw,
    const float* __restrict__ Uq, float* __restrict__ bpart) {
  __shared__ _Float16 sG[WAVES_PER_BLOCK][ROWS_PER_WAVE * E_];
  __shared__ _Float16 sHn[WAVES_PER_BLOCK][ROWS_PER_WAVE * E_];
  __shared__ float sPart[WAVES_PER_BLOCK][K_];

  const _Float16* w1h = hbase + H_W1;
  const _Float16* w2h = hbase + H_W2;
  const _Float16* wwh = hbase + H_WW;

  const int lane = threadIdx.x & 31;
  const int wave = threadIdx.x >> 5;
  const int tile = blockIdx.x * WAVES_PER_BLOCK + wave;
  const bool active = (tile < NTILES);
  const int hi = (lane >> 4) & 1;
  const int lx = lane & 15;
  const int r0 = tile * ROWS_PER_WAVE;

  // ---- stage A: layer1 (X @ w1^T + b1, relu) -> sG ----
  if (active) {
    v16h aX = loadA_X(X, r0 + lx, hi);
#pragma unroll
    for (int n = 0; n < 4; ++n) {
      v16h B = loadB(w1h + (16 * n + lx) * F_ + (hi ? 16 : 0));
      v8f c = {};
      c = wmma_f16(aX, B, c);
      float bn = b1[16 * n + lx];
#pragma unroll
      for (int r = 0; r < 8; ++r) {
        float v = c[r] + bn;
        v = v > 0.f ? v : 0.f;
        sG[wave][(r + (hi ? 8 : 0)) * E_ + 16 * n + lx] = (_Float16)v;
      }
    }
  }
  __syncthreads();

  v8f acc[4];
  // ---- stage B: layer2 + LayerNorm -> sHn ----
  if (active) {
    v16h hA0 = loadA_lds(sG[wave], lx, 0, hi);
    v16h hA1 = loadA_lds(sG[wave], lx, 1, hi);
#pragma unroll
    for (int n = 0; n < 4; ++n) {
      v16h B0 = loadB(w2h + (16 * n + lx) * E_ + 0 + (hi ? 16 : 0));
      v16h B1 = loadB(w2h + (16 * n + lx) * E_ + 32 + (hi ? 16 : 0));
      v8f c = {};
      c = wmma_f16(hA0, B0, c);
      c = wmma_f16(hA1, B1, c);
      float bn = b2[16 * n + lx];
#pragma unroll
      for (int r = 0; r < 8; ++r) c[r] += bn;
      acc[n] = c;
    }
    float g[4], bt[4];
#pragma unroll
    for (int n = 0; n < 4; ++n) { g[n] = gam[16 * n + lx]; bt[n] = bet[16 * n + lx]; }
#pragma unroll
    for (int r = 0; r < 8; ++r) {
      float s1 = 0.f, s2 = 0.f;
#pragma unroll
      for (int n = 0; n < 4; ++n) { float v = acc[n][r]; s1 += v; s2 += v * v; }
      s1 = half_reduce(s1);
      s2 = half_reduce(s2);
      float mu = s1 * (1.f / E_);
      float var = s2 * (1.f / E_) - mu * mu;
      float rs = rsqrtf(var + LN_EPS);
#pragma unroll
      for (int n = 0; n < 4; ++n) {
        float v = (acc[n][r] - mu) * rs * g[n] + bt[n];
        acc[n][r] = v;
        sHn[wave][(r + (hi ? 8 : 0)) * E_ + 16 * n + lx] = (_Float16)v;
      }
    }
  }
  __syncthreads();

  // ---- stage C: Wh, scores, per-wave exp-sum ----
  if (active) {
    v16h nA0 = loadA_lds(sHn[wave], lx, 0, hi);
    v16h nA1 = loadA_lds(sHn[wave], lx, 1, hi);
    float wh[4][8];
#pragma unroll
    for (int n = 0; n < 4; ++n) {
      v16h B0 = loadB(wwh + (16 * n + lx) * E_ + 0 + (hi ? 16 : 0));
      v16h B1 = loadB(wwh + (16 * n + lx) * E_ + 32 + (hi ? 16 : 0));
      v8f c = {};
      c = wmma_f16(nA0, B0, c);
      c = wmma_f16(nA1, B1, c);
      float bn = Wb[16 * n + lx];
#pragma unroll
      for (int r = 0; r < 8; ++r) wh[n][r] = c[r] + bn;
    }
    float vwr[4], uqr[K_][4];
#pragma unroll
    for (int n = 0; n < 4; ++n) vwr[n] = vw[16 * n + lx];
#pragma unroll
    for (int k = 0; k < K_; ++k)
#pragma unroll
      for (int n = 0; n < 4; ++n) uqr[k][n] = Uq[k * E_ + 16 * n + lx];

#pragma unroll
    for (int k = 0; k < K_; ++k) {
      float psum = 0.f;
#pragma unroll
      for (int r = 0; r < 8; ++r) {
        float t = 0.f;
#pragma unroll
        for (int n = 0; n < 4; ++n) t += fast_tanh(wh[n][r] + uqr[k][n]) * vwr[n];
        t = half_reduce(t);                 // score for row r0 + r + 8*hi
        float e = __expf(t);
        if (lx == 0) psum += e;
      }
      psum += __shfl_xor(psum, 16, 32);
      if (lane == 0) sPart[wave][k] = psum;
    }
  } else {
    if (lane < K_) sPart[wave][lane] = 0.f;
  }
  __syncthreads();

  // ---- stage D: fixed-order block reduce ----
  if (wave == 0 && lane < K_) {
    float s = 0.f;
#pragma unroll
    for (int w = 0; w < WAVES_PER_BLOCK; ++w) s += sPart[w][lane];
    bpart[blockIdx.x * K_ + lane] = s;
  }
}

__global__ void denom_kernel(const float* __restrict__ bpart, float* __restrict__ den) {
  int k = threadIdx.x;
  if (k < K_) {
    float s = 0.f;
    for (int b = 0; b < NBLK; ++b) s += bpart[b * K_ + k];
    den[k] = s;
  }
}

// Pass 2: recompute H + scores, emit weights, accumulate Zk block-partials
__global__ __launch_bounds__(256) void pass2_kernel(
    const float* __restrict__ X, const _Float16* __restrict__ hbase,
    const float* __restrict__ b1, const float* __restrict__ b2,
    const float* __restrict__ gam, const float* __restrict__ bet,
    const float* __restrict__ Wb, const float* __restrict__ vw,
    const float* __restrict__ Uq, const float* __restrict__ den,
    float* __restrict__ wout,    // d_out + 64 : weights[k*L + l]
    float* __restrict__ zpart) { // [NBLK][K*E]
  __shared__ _Float16 sG[WAVES_PER_BLOCK][ROWS_PER_WAVE * E_];
  __shared__ _Float16 sHn[WAVES_PER_BLOCK][ROWS_PER_WAVE * E_];
  __shared__ float sZ[WAVES_PER_BLOCK][K_ * E_];

  const _Float16* w1h = hbase + H_W1;
  const _Float16* w2h = hbase + H_W2;
  const _Float16* wwh = hbase + H_WW;

  const int lane = threadIdx.x & 31;
  const int wave = threadIdx.x >> 5;
  const int tile = blockIdx.x * WAVES_PER_BLOCK + wave;
  const bool active = (tile < NTILES);
  const int hi = (lane >> 4) & 1;
  const int lx = lane & 15;
  const int r0 = tile * ROWS_PER_WAVE;

  // ---- stage A: layer1 -> sG ----
  if (active) {
    v16h aX = loadA_X(X, r0 + lx, hi);
#pragma unroll
    for (int n = 0; n < 4; ++n) {
      v16h B = loadB(w1h + (16 * n + lx) * F_ + (hi ? 16 : 0));
      v8f c = {};
      c = wmma_f16(aX, B, c);
      float bn = b1[16 * n + lx];
#pragma unroll
      for (int r = 0; r < 8; ++r) {
        float v = c[r] + bn;
        v = v > 0.f ? v : 0.f;
        sG[wave][(r + (hi ? 8 : 0)) * E_ + 16 * n + lx] = (_Float16)v;
      }
    }
  }
  __syncthreads();

  v8f acc[4];
  // ---- stage B: layer2 + LN -> sHn (and keep C-layout H in acc) ----
  if (active) {
    v16h hA0 = loadA_lds(sG[wave], lx, 0, hi);
    v16h hA1 = loadA_lds(sG[wave], lx, 1, hi);
#pragma unroll
    for (int n = 0; n < 4; ++n) {
      v16h B0 = loadB(w2h + (16 * n + lx) * E_ + 0 + (hi ? 16 : 0));
      v16h B1 = loadB(w2h + (16 * n + lx) * E_ + 32 + (hi ? 16 : 0));
      v8f c = {};
      c = wmma_f16(hA0, B0, c);
      c = wmma_f16(hA1, B1, c);
      float bn = b2[16 * n + lx];
#pragma unroll
      for (int r = 0; r < 8; ++r) c[r] += bn;
      acc[n] = c;
    }
    float g[4], bt[4];
#pragma unroll
    for (int n = 0; n < 4; ++n) { g[n] = gam[16 * n + lx]; bt[n] = bet[16 * n + lx]; }
#pragma unroll
    for (int r = 0; r < 8; ++r) {
      float s1 = 0.f, s2 = 0.f;
#pragma unroll
      for (int n = 0; n < 4; ++n) { float v = acc[n][r]; s1 += v; s2 += v * v; }
      s1 = half_reduce(s1);
      s2 = half_reduce(s2);
      float mu = s1 * (1.f / E_);
      float var = s2 * (1.f / E_) - mu * mu;
      float rs = rsqrtf(var + LN_EPS);
#pragma unroll
      for (int n = 0; n < 4; ++n) {
        float v = (acc[n][r] - mu) * rs * g[n] + bt[n];
        acc[n][r] = v;
        sHn[wave][(r + (hi ? 8 : 0)) * E_ + 16 * n + lx] = (_Float16)v;
      }
    }
  }
  __syncthreads();

  // ---- stage C: Wh, weights, Zk accumulation ----
  if (active) {
    v16h nA0 = loadA_lds(sHn[wave], lx, 0, hi);
    v16h nA1 = loadA_lds(sHn[wave], lx, 1, hi);
    float wh[4][8];
#pragma unroll
    for (int n = 0; n < 4; ++n) {
      v16h B0 = loadB(wwh + (16 * n + lx) * E_ + 0 + (hi ? 16 : 0));
      v16h B1 = loadB(wwh + (16 * n + lx) * E_ + 32 + (hi ? 16 : 0));
      v8f c = {};
      c = wmma_f16(nA0, B0, c);
      c = wmma_f16(nA1, B1, c);
      float bn = Wb[16 * n + lx];
#pragma unroll
      for (int r = 0; r < 8; ++r) wh[n][r] = c[r] + bn;
    }
    float vwr[4], uqr[K_][4], dinv[K_];
#pragma unroll
    for (int n = 0; n < 4; ++n) vwr[n] = vw[16 * n + lx];
#pragma unroll
    for (int k = 0; k < K_; ++k) {
      dinv[k] = 1.f / den[k];
#pragma unroll
      for (int n = 0; n < 4; ++n) uqr[k][n] = Uq[k * E_ + 16 * n + lx];
    }

#pragma unroll
    for (int k = 0; k < K_; ++k) {
      float zacc[4] = {0.f, 0.f, 0.f, 0.f};
#pragma unroll
      for (int r = 0; r < 8; ++r) {
        float t = 0.f;
#pragma unroll
        for (int n = 0; n < 4; ++n) t += fast_tanh(wh[n][r] + uqr[k][n]) * vwr[n];
        t = half_reduce(t);
        float w = __expf(t) * dinv[k];      // softmax weight for this row
        if (lx == 0) wout[k * L_ + r0 + r + (hi ? 8 : 0)] = w;
#pragma unroll
        for (int n = 0; n < 4; ++n) zacc[n] += w * acc[n][r];
      }
#pragma unroll
      for (int n = 0; n < 4; ++n) {
        float z = zacc[n] + __shfl_xor(zacc[n], 16, 32);
        if (!hi) sZ[wave][k * E_ + 16 * n + lx] = z;
      }
    }
  } else {
    for (int t = lane; t < K_ * E_; t += 32) sZ[wave][t] = 0.f;
  }
  __syncthreads();

  // ---- stage D: fixed-order block reduce of Zk partials ----
  {
    int t = threadIdx.x;  // 0..255 == K_*E_
    float s = 0.f;
#pragma unroll
    for (int w = 0; w < WAVES_PER_BLOCK; ++w) s += sZ[w][t];
    zpart[blockIdx.x * (K_ * E_) + t] = s;
  }
}

__global__ void zred_kernel(const float* __restrict__ zpart, float* __restrict__ Z) {
  int t = threadIdx.x;  // 256 threads
  float s = 0.f;
  for (int b = 0; b < NBLK; ++b) s += zpart[b * (K_ * E_) + t];
  Z[t] = s;
}

__global__ __launch_bounds__(64) void rho_kernel(
    const float* __restrict__ Z, const float* __restrict__ rw1,
    const float* __restrict__ rb1, const float* __restrict__ rw2,
    const float* __restrict__ rb2, float* __restrict__ zout) {
  __shared__ float hid[HID_];
  int t = threadIdx.x;
  float s = rb1[t];
  for (int i = 0; i < K_ * E_; ++i) s += Z[i] * rw1[t * (K_ * E_) + i];
  hid[t] = fmaxf(s, 0.f);
  __syncthreads();
  float o = rb2[t];
  for (int j = 0; j < HID_; ++j) o += hid[j] * rw2[t * HID_ + j];
  zout[t] = o;
}

// ---------------------------------------------------------------- launch ----

extern "C" void kernel_launch(void* const* d_in, const int* in_sizes, int n_in,
                              void* d_out, int out_size, void* d_ws, size_t ws_size,
                              hipStream_t stream) {
  const float* X     = (const float*)d_in[0];
  const float* w1    = (const float*)d_in[1];
  const float* b1    = (const float*)d_in[2];
  const float* w2    = (const float*)d_in[3];
  const float* b2    = (const float*)d_in[4];
  const float* gamma = (const float*)d_in[5];
  const float* beta  = (const float*)d_in[6];
  const float* Ww    = (const float*)d_in[7];
  const float* Wb    = (const float*)d_in[8];
  const float* Uw    = (const float*)d_in[9];
  const float* vw    = (const float*)d_in[10];
  const float* seeds = (const float*)d_in[11];
  const float* rw1   = (const float*)d_in[12];
  const float* rb1   = (const float*)d_in[13];
  const float* rw2   = (const float*)d_in[14];
  const float* rb2   = (const float*)d_in[15];

  float* out = (float*)d_out;         // [0..63] = z, [64 ..] = weights[k*L + l]
  float* ws  = (float*)d_ws;
  _Float16* hbase = (_Float16*)(ws + O_HALF);

  prep_kernel<<<1, 256, 0, stream>>>(seeds, Uw, w1, w2, Ww, ws + O_UQ, hbase);
  pass1_kernel<<<NBLK, 256, 0, stream>>>(X, hbase, b1, b2, gamma, beta, Wb, vw,
                                         ws + O_UQ, ws + O_BPART);
  denom_kernel<<<1, 64, 0, stream>>>(ws + O_BPART, ws + O_DEN);
  pass2_kernel<<<NBLK, 256, 0, stream>>>(X, hbase, b1, b2, gamma, beta, Wb, vw,
                                         ws + O_UQ, ws + O_DEN, out + OUT_,
                                         ws + O_ZPART);
  zred_kernel<<<1, 256, 0, stream>>>(ws + O_ZPART, ws + O_Z);
  rho_kernel<<<1, 64, 0, stream>>>(ws + O_Z, rw1, rb1, rw2, rb2, out);
}
